// WindowAttention_47588237640400
// MI455X (gfx1250) — matrix operand order repaired and checked by reading, not compiled, not run on hardware
//
#include <hip/hip_runtime.h>
#include <hip/hip_bf16.h>

typedef __attribute__((ext_vector_type(16))) __bf16 v16bf;
typedef __attribute__((ext_vector_type(8)))  __bf16 v8bf;
typedef __attribute__((ext_vector_type(8)))  float  v8f;

#define DIMC   256
#define NQKV   768
#define NTOK   64
#define HEADS_ 8
#define HD     32
#define NEG_BIG -1e30f

// RNE float->bf16 (used only in the one-time weight swizzle).
__device__ __forceinline__ __bf16 f2bf_rne(float f) {
    union { float f; unsigned u; } a; a.f = f;
    unsigned r = a.u + 0x7FFFu + ((a.u >> 16) & 1u);
    unsigned short s = (unsigned short)(r >> 16);
    __bf16 o; __builtin_memcpy(&o, &s, sizeof(o));
    return o;
}

// Truncating bf16 store for activations: high half of the f32 bit pattern.
// Lowers to ds_store_b16_d16_hi (or lshr+b16) -- ~0-1 VALU per element.
__device__ __forceinline__ void st_bf16(__bf16* p, float f) {
    union { float f; unsigned u; } a; a.f = f;
    *(unsigned short*)p = (unsigned short)(a.u >> 16);
}

// Pack two floats' bf16 truncations into one dword: {hi(b), hi(a)}.
__device__ __forceinline__ unsigned pk_bf16(float fa, float fb) {
    union { float f; unsigned u; } a, b; a.f = fa; b.f = fb;
    return (b.u & 0xFFFF0000u) | (a.u >> 16);
}

__device__ __forceinline__ v8f wmma_bf16(v16bf a, v16bf b, v8f c) {
    return __builtin_amdgcn_wmma_f32_16x16x32_bf16(
        false, a, false, b, (short)0, c, false, false);
}

// A-operand tile from row-major bf16: 16 contiguous elements (two 8-elem chunks).
__device__ __forceinline__ v16bf load_a(const __bf16* p) {
    union { v16bf v; v8bf h[2]; } u;
    u.h[0] = *(const v8bf*)(p);
    u.h[1] = *(const v8bf*)(p + 8);
    return u.v;
}

// ---------------------------------------------------------------------------
// Weight pre-swizzle: fp32 row-major [K][N] -> bf16 in WMMA-B tile order.
// ---------------------------------------------------------------------------
__global__ void swizzle_weights(const float* __restrict__ wqkv,
                                const float* __restrict__ wout,
                                __bf16* __restrict__ sqkv,
                                __bf16* __restrict__ soutw) {
    int e = blockIdx.x * blockDim.x + threadIdx.x;   // 0 .. 262143
    const int EQ = DIMC * NQKV;
    int which = (e >= EQ) ? 1 : 0;
    int idx   = which ? (e - EQ) : e;
    int N     = which ? DIMC : NQKV;
    int nnt   = N >> 4;
    int j     = idx & 15;
    int lane  = (idx >> 4) & 31;
    int tile  = idx >> 9;
    int nt    = tile % nnt;
    int kt    = tile / nnt;
    int k     = kt * 32 + ((lane >> 4) << 4) + j;
    int n     = nt * 16 + (lane & 15);
    if (which) soutw[idx] = f2bf_rne(wout[k * DIMC + n]);
    else       sqkv[idx]  = f2bf_rne(wqkv[k * NQKV + n]);
}

// ---------------------------------------------------------------------------
// Fully fused shifted-window attention: one workgroup per (batch, window).
// 256 threads = 8 wave32; wave id == head id in the attention stage.
// ---------------------------------------------------------------------------
__global__ __launch_bounds__(256, 1)
void swin_fused(const float* __restrict__ x,
                const float* __restrict__ pos,
                const float* __restrict__ bout,
                const __bf16* __restrict__ sqkv,
                const __bf16* __restrict__ soutw,
                float* __restrict__ out) {
    __shared__ __bf16 sXO[NTOK * DIMC];          // 32KB: X in, attn-out later
    __shared__ __bf16 sQ [NTOK * DIMC];          // 32KB
    __shared__ __bf16 sK [NTOK * DIMC];          // 32KB
    __shared__ __bf16 sVt[DIMC * NTOK];          // 32KB transposed [ch][tok]
    __shared__ __bf16 sP [HEADS_ * NTOK * NTOK]; // 64KB probs
    __shared__ float  sPos[225];

    const int tid  = threadIdx.x;
    const int lane = tid & 31;
    const int wid  = tid >> 5;
    const int l15  = lane & 15;
    const int lhi  = lane >> 4;

    const int bi = blockIdx.x >> 8;
    const int wy = (blockIdx.x >> 4) & 15;
    const int wx = blockIdx.x & 15;

    if (tid < 225) sPos[tid] = pos[tid];

    // ---- Stage 1: rolled x window -> bf16 LDS, lane-coalesced float4 ----
    {
        const long xbase = (long)bi * 128 * 128 * DIMC;
#pragma unroll
        for (int it = 0; it < 16; ++it) {
            int flat = it * 1024 + tid * 4;     // float idx in [64 x 256]
            int t  = flat >> 8;                 // token
            int cb = flat & 255;                // channel
            int iy = (wy * 8 + (t >> 3) + 4) & 127;   // roll(-4)
            int ix = (wx * 8 + (t & 7) + 4) & 127;
            float4 f = *(const float4*)(x + xbase + ((long)iy * 128 + ix) * DIMC + cb);
            unsigned2_hack:;
            unsigned d0 = pk_bf16(f.x, f.y);
            unsigned d1 = pk_bf16(f.z, f.w);
            *(uint2*)(sXO + flat) = make_uint2(d0, d1);
        }
    }
    __syncthreads();

    // ---- Stage 2: QKV GEMM  C[64,768] = Xbf[64,256] @ w_qkv ----
    for (int j = 0; j < 6; ++j) {
        int ng = wid * 6 + j;                       // n-tile 0..47
        v8f acc[4] = {};
#pragma unroll
        for (int kt = 0; kt < 8; ++kt) {
            v16bf B = *(const v16bf*)(sqkv + (((long)kt * 48 + ng) * 32 + lane) * 16);
#pragma unroll
            for (int mt = 0; mt < 4; ++mt) {
                v16bf A = load_a(sXO + (mt * 16 + l15) * DIMC + kt * 32 + lhi * 16);
                acc[mt] = wmma_bf16(A, B, acc[mt]);
            }
        }
        int col = ng * 16;                          // 0..767
#pragma unroll
        for (int mt = 0; mt < 4; ++mt) {
#pragma unroll
            for (int r = 0; r < 8; ++r) {
                int m = mt * 16 + lhi * 8 + r;      // C layout: M = r + 8*hi
                float v = acc[mt][r];
                if (col < 256)      st_bf16(&sQ[m * DIMC + col + l15], v);
                else if (col < 512) st_bf16(&sK[m * DIMC + (col - 256) + l15], v);
                else                st_bf16(&sVt[((col - 512) + l15) * NTOK + m], v);
            }
        }
    }
    __syncthreads();

    // ---- Stage 3: attention, one head per wave ----
    {
        const int h = wid;
        const __bf16* Qb = sQ + h * HD;
        const __bf16* Kb = sK + h * HD;
        __bf16* Pb = sP + h * (NTOK * NTOK);
        const bool mUL = (wy == 15);
        const bool mLR = (wx == 15);

#pragma unroll
        for (int mt = 0; mt < 4; ++mt) {
            v16bf A = load_a(Qb + (mt * 16 + l15) * DIMC + lhi * 16);
            v8f s[4];
#pragma unroll
            for (int nt = 0; nt < 4; ++nt) {
                v16bf B = *(const v16bf*)(Kb + (nt * 16 + l15) * DIMC + lhi * 16);
                v8f z = {};
                s[nt] = wmma_bf16(A, B, z);
            }
#pragma unroll
            for (int r = 0; r < 8; ++r) {
                int m = mt * 16 + lhi * 8 + r;
                float vv[4];
#pragma unroll
                for (int nt = 0; nt < 4; ++nt) {
                    int n = nt * 16 + l15;
                    float sv = s[nt][r] * 0.17677669529663687f;
                    int r0 = (n >> 3) - (m >> 3) + 7;
                    int r1 = (n & 7) - (m & 7) + 7;
                    sv += sPos[r0 * 15 + r1];
                    if (mUL && ((m >= 32) != (n >= 32)))           sv = NEG_BIG;
                    if (mLR && (((m & 7) >= 4) != ((n & 7) >= 4))) sv = NEG_BIG;
                    vv[nt] = sv;
                }
                float mx = fmaxf(fmaxf(vv[0], vv[1]), fmaxf(vv[2], vv[3]));
#pragma unroll
                for (int d = 1; d < 16; d <<= 1) mx = fmaxf(mx, __shfl_xor(mx, d, 32));
                float e0 = __expf(vv[0] - mx), e1 = __expf(vv[1] - mx);
                float e2 = __expf(vv[2] - mx), e3 = __expf(vv[3] - mx);
                float sum = e0 + e1 + e2 + e3;
#pragma unroll
                for (int d = 1; d < 16; d <<= 1) sum += __shfl_xor(sum, d, 32);
                float inv = 1.0f / sum;
                st_bf16(&Pb[m * NTOK +  0 + l15], e0 * inv);
                st_bf16(&Pb[m * NTOK + 16 + l15], e1 * inv);
                st_bf16(&Pb[m * NTOK + 32 + l15], e2 * inv);
                st_bf16(&Pb[m * NTOK + 48 + l15], e3 * inv);
            }
        }

        // O[64,32] = P[64,64] @ V[64,32]  (V transposed -> contiguous B)
#pragma unroll
        for (int mt = 0; mt < 4; ++mt) {
            v8f o[2] = {};
#pragma unroll
            for (int ks = 0; ks < 2; ++ks) {
                v16bf A = load_a(Pb + (mt * 16 + l15) * NTOK + ks * 32 + lhi * 16);
#pragma unroll
                for (int n2 = 0; n2 < 2; ++n2) {
                    v16bf B = *(const v16bf*)(sVt + (h * HD + n2 * 16 + l15) * NTOK
                                              + ks * 32 + lhi * 16);
                    o[n2] = wmma_bf16(A, B, o[n2]);
                }
            }
#pragma unroll
            for (int n2 = 0; n2 < 2; ++n2)
#pragma unroll
                for (int r = 0; r < 8; ++r) {
                    int m = mt * 16 + lhi * 8 + r;
                    st_bf16(&sXO[m * DIMC + h * HD + n2 * 16 + l15], o[n2][r]);
                }
        }
    }
    __syncthreads();

    // ---- Stage 4: output GEMM [64,256] = O @ w_out, + bias, roll(+4) ----
    for (int j = 0; j < 2; ++j) {
        int ng = wid * 2 + j;                       // 0..15
        v8f acc[4] = {};
#pragma unroll
        for (int kt = 0; kt < 8; ++kt) {
            v16bf B = *(const v16bf*)(soutw + (((long)kt * 16 + ng) * 32 + lane) * 16);
#pragma unroll
            for (int mt = 0; mt < 4; ++mt) {
                v16bf A = load_a(sXO + (mt * 16 + l15) * DIMC + kt * 32 + lhi * 16);
                acc[mt] = wmma_bf16(A, B, acc[mt]);
            }
        }
        int ch = ng * 16 + l15;
        float bb = bout[ch];
#pragma unroll
        for (int mt = 0; mt < 4; ++mt) {
#pragma unroll
            for (int r = 0; r < 8; ++r) {
                int m = mt * 16 + lhi * 8 + r;
                int oy = (wy * 8 + (m >> 3) + 4) & 127;   // roll(+4)
                int ox = (wx * 8 + (m & 7) + 4) & 127;
                out[(((long)bi * 128 + oy) * 128 + ox) * DIMC + ch] = acc[mt][r] + bb;
            }
        }
    }
}

extern "C" void kernel_launch(void* const* d_in, const int* in_sizes, int n_in,
                              void* d_out, int out_size, void* d_ws, size_t ws_size,
                              hipStream_t stream) {
    const float* x    = (const float*)d_in[0];
    const float* wqkv = (const float*)d_in[1];
    const float* pos  = (const float*)d_in[2];
    const float* wout = (const float*)d_in[3];
    const float* bout = (const float*)d_in[4];
    float* out = (float*)d_out;

    __bf16* sqkv  = (__bf16*)d_ws;            // 256*768 bf16
    __bf16* soutw = sqkv + DIMC * NQKV;       // 256*256 bf16

    swizzle_weights<<<1024, 256, 0, stream>>>(wqkv, wout, sqkv, soutw);
    swin_fused<<<2048, 256, 0, stream>>>(x, pos, bout, sqkv, soutw, out);
}
